// LSTMSeq2SeqDecoder_9036611191294
// MI455X (gfx1250) — matrix operand order repaired and checked by reading, not compiled, not run on hardware
//
#include <hip/hip_runtime.h>

// ---------------- problem constants ----------------
constexpr int Vn = 32000, En = 512, Hn = 512;
constexpr int Bn = 32, Tn = 64, Sn = 128;
constexpr int G4 = 4 * Hn;          // 2048 gate width
constexpr int K0n = En + Hn + Hn;   // 1536: [x | feed | h0]
constexpr int K1n = Hn + Hn;        // 1024: [h0 | h1]

typedef __attribute__((ext_vector_type(16))) __bf16 v16bf;
typedef __attribute__((ext_vector_type(8)))  float  v8f;

// ---------------- helpers ----------------
__device__ __forceinline__ unsigned short f2bf(float f) {
  union { float f; unsigned u; } v; v.f = f;
  unsigned r = v.u + 0x7FFFu + ((v.u >> 16) & 1u);
  return (unsigned short)(r >> 16);
}
__device__ __forceinline__ float sigf(float x) { return 1.f / (1.f + __expf(-x)); }

// A fragment: 16x32 bf16 (MxK), row-major source, per ISA 16-bit A layout:
// lanes 0-15 row M=lane, K = {0..7,16..23}; lanes 16-31 same rows, K+8.
__device__ __forceinline__ v16bf load_a_frag(const __bf16* A, int lda) {
  const int lane = threadIdx.x & 31;
  const int half = lane >> 4, row = lane & 15;
  const __bf16* p = A + (size_t)row * lda + half * 8;
  v16bf a;
#pragma unroll
  for (int g = 0; g < 2; ++g)
#pragma unroll
    for (int i = 0; i < 8; ++i)
      a[g * 8 + i] = p[g * 16 + i];
  return a;
}

// B fragment: B[k][n] = W[n][k] (W row-major [N,K]); lanes 0-15 K={0..7,16..23},
// lanes 16-31 same K+8 (matches A layout), N = lane%16.
__device__ __forceinline__ v16bf load_bt_frag(const __bf16* W, int ldw) {
  const int lane = threadIdx.x & 31;
  const int half = lane >> 4, col = lane & 15;
  const __bf16* p = W + (size_t)col * ldw + half * 8;
  v16bf b;
#pragma unroll
  for (int i = 0; i < 16; ++i) b[i] = p[(i >> 3) * 16 + (i & 7)];
  return b;
}

// ---------------- generic bf16 WMMA GEMM:  out = A[M,K] @ W[N,K]^T (+bias)(+tanh)
// Each wave computes a 16 x (16*NT) tile, accumulating f32 via v_wmma_f32_16x16x32_bf16.
template <int NT, bool TANH>
__global__ void gemm_bf16_wmma(const __bf16* __restrict__ A, int lda,
                               const __bf16* __restrict__ W, int ldw,
                               const float* __restrict__ bias,
                               float* __restrict__ outF, int ldoF,
                               unsigned short* __restrict__ outB, int ldoB,
                               unsigned short* __restrict__ outB2, int ldoB2,
                               int M, int N, int K) {
  const int wave   = blockIdx.x * (blockDim.x >> 5) + (threadIdx.x >> 5);
  const int tilesM = M >> 4;
  const int tilesG = N / (16 * NT);
  if (wave >= tilesM * tilesG) return;          // wave-uniform: EXEC stays all-ones
  const int m0 = (wave % tilesM) << 4;
  const int n0 = (wave / tilesM) * (16 * NT);

  v8f acc[NT] = {};
  const __bf16* Abase = A + (size_t)m0 * lda;
  for (int k0 = 0; k0 < K; k0 += 32) {
    v16bf a = load_a_frag(Abase + k0, lda);
#pragma unroll
    for (int j = 0; j < NT; ++j) {
      v16bf b = load_bt_frag(W + (size_t)(n0 + 16 * j) * ldw + k0, ldw);
      acc[j] = __builtin_amdgcn_wmma_f32_16x16x32_bf16(
          false, a, false, b, (short)0, acc[j], false, false);
    }
  }

  const int lane = threadIdx.x & 31;
  const int half = lane >> 4, col = lane & 15;
#pragma unroll
  for (int j = 0; j < NT; ++j) {
    const int n = n0 + 16 * j + col;
    const float bv = bias ? bias[n] : 0.f;
#pragma unroll
    for (int r = 0; r < 8; ++r) {
      const int m = m0 + r + half * 8;          // C/D layout: M = r + 8*(lane/16)
      float v = acc[j][r] + bv;
      if (TANH) v = tanhf(v);
      if (outF)  outF[(size_t)m * ldoF + n]  = v;
      if (outB)  outB[(size_t)m * ldoB + n]  = f2bf(v);
      if (outB2) outB2[(size_t)m * ldoB2 + n] = f2bf(v);
    }
  }
}

// ---------------- setup / pointwise kernels ----------------
__global__ void k_zero(unsigned int* p, int n) {
  int i = blockIdx.x * blockDim.x + threadIdx.x;
  if (i < n) p[i] = 0u;
}
__global__ void k_cvt(const float* __restrict__ s, unsigned short* __restrict__ d, int n) {
  int i = blockIdx.x * blockDim.x + threadIdx.x;
  if (i < n) d[i] = f2bf(s[i]);
}
__global__ void k_bias_sum(const float* a, const float* b, float* o, int n) {
  int i = blockIdx.x * blockDim.x + threadIdx.x;
  if (i < n) o[i] = a[i] + b[i];
}
// W0cat[n, 0:1024] = w_ih0[n,:], W0cat[n, 1024:1536] = w_hh0[n,:]
__global__ void k_build_w0(const float* __restrict__ wih, const float* __restrict__ whh,
                           unsigned short* __restrict__ Wc) {
  int i = blockIdx.x * blockDim.x + threadIdx.x;
  if (i >= G4 * K0n) return;
  int n = i / K0n, k = i % K0n;
  float v = (k < 1024) ? wih[(size_t)n * 1024 + k] : whh[(size_t)n * 512 + (k - 1024)];
  Wc[i] = f2bf(v);
}
// W1cat[n, 0:512] = w_ih1[n,:], W1cat[n, 512:1024] = w_hh1[n,:]
__global__ void k_build_w1(const float* __restrict__ wih, const float* __restrict__ whh,
                           unsigned short* __restrict__ Wc) {
  int i = blockIdx.x * blockDim.x + threadIdx.x;
  if (i >= G4 * K1n) return;
  int n = i / K1n, k = i % K1n;
  float v = (k < 512) ? wih[(size_t)n * 512 + k] : whh[(size_t)n * 512 + (k - 512)];
  Wc[i] = f2bf(v);
}
__global__ void k_embed_gather(const int* __restrict__ tok, const float* __restrict__ emb,
                               unsigned short* __restrict__ xbf) {
  int i = blockIdx.x * blockDim.x + threadIdx.x;
  if (i >= Bn * Tn * En) return;
  int bt = i >> 9, j = i & 511;
  xbf[i] = f2bf(emb[(size_t)tok[bt] * En + j]);
}
// copy x_t into A0[:, 0:512] (row stride K0n)
__global__ void k_set_xt(const unsigned short* __restrict__ xbf,
                         unsigned short* __restrict__ A0, int t) {
  int i = blockIdx.x * blockDim.x + threadIdx.x;
  if (i >= Bn * En) return;
  int b = i >> 9, j = i & 511;
  A0[(size_t)b * K0n + j] = xbf[((size_t)b * Tn + t) * En + j];
}
// torch gate order i,f,g,o; writes h (bf16) to two destinations
__global__ void k_lstm(const float* __restrict__ gates, float* __restrict__ c,
                       unsigned short* __restrict__ h1, int ld1,
                       unsigned short* __restrict__ h2, int ld2) {
  int i = blockIdx.x * blockDim.x + threadIdx.x;
  if (i >= Bn * Hn) return;
  int b = i >> 9, j = i & 511;
  const float* g = gates + (size_t)b * G4;
  float ig = sigf(g[j]);
  float fg = sigf(g[512 + j]);
  float gg = tanhf(g[1024 + j]);
  float og = sigf(g[1536 + j]);
  float cc = fg * c[i] + ig * gg;
  c[i] = cc;
  unsigned short hb = f2bf(og * tanhf(cc));
  h1[(size_t)b * ld1 + j] = hb;
  h2[(size_t)b * ld2 + j] = hb;
}
// fused scores -> masked softmax -> context; one workgroup per batch row.
// float4 loads force b128 vmem for the encoder streams (L2-resident, 8MB).
__global__ void k_attn(const float* __restrict__ enc, const int* __restrict__ mask,
                       const float* __restrict__ q, unsigned short* __restrict__ Aatt) {
  const int b = blockIdx.x, tid = threadIdx.x;   // 128 threads, one per source pos
  __shared__ float prob[Sn];
  __shared__ float red[Sn];
  const float* eb = enc + (size_t)b * Sn * Hn;
  const float* qb = q + (size_t)b * Hn;          // block-uniform -> scalar loads
  const float4* er4 = (const float4*)(eb + (size_t)tid * Hn);
  const float4* qb4 = (const float4*)qb;
  float s = 0.f;
#pragma unroll 4
  for (int k = 0; k < Hn / 4; ++k) {
    float4 e = er4[k], w = qb4[k];
    s += e.x * w.x + e.y * w.y + e.z * w.z + e.w * w.w;
  }
  if (mask[b * Sn + tid] == 0) s = -1e4f;
  red[tid] = s; __syncthreads();
  for (int o = Sn / 2; o > 0; o >>= 1) {
    if (tid < o) red[tid] = fmaxf(red[tid], red[tid + o]);
    __syncthreads();
  }
  float mx = red[0]; __syncthreads();
  float e = __expf(s - mx);
  red[tid] = e; __syncthreads();
  for (int o = Sn / 2; o > 0; o >>= 1) {
    if (tid < o) red[tid] += red[tid + o];
    __syncthreads();
  }
  float inv = 1.f / red[0];
  prob[tid] = e * inv;
  __syncthreads();
  // ctx: threads cover 128 h-columns at a time (coalesced), 4 h per thread
  for (int h0 = 0; h0 < Hn; h0 += Sn) {
    const int h = h0 + tid;
    float acc = 0.f;
#pragma unroll 4
    for (int s2 = 0; s2 < Sn; ++s2) acc += prob[s2] * eb[(size_t)s2 * Hn + h];
    Aatt[(size_t)b * K1n + h] = f2bf(acc);      // ctx -> Aatt[:, 0:512]
  }
}

// ---------------- host ----------------
#define CDIV(a, b) (((a) + (b) - 1) / (b))

extern "C" void kernel_launch(void* const* d_in, const int* in_sizes, int n_in,
                              void* d_out, int out_size, void* d_ws, size_t ws_size,
                              hipStream_t stream) {
  const int*   tokens   = (const int*)d_in[0];
  const float* enc      = (const float*)d_in[1];
  const int*   mask     = (const int*)d_in[2];
  const float* embed    = (const float*)d_in[3];
  const float* w_ih0    = (const float*)d_in[4];
  const float* w_hh0    = (const float*)d_in[5];
  const float* b_ih0    = (const float*)d_in[6];
  const float* b_hh0    = (const float*)d_in[7];
  const float* w_ih1    = (const float*)d_in[8];
  const float* w_hh1    = (const float*)d_in[9];
  const float* b_ih1    = (const float*)d_in[10];
  const float* b_hh1    = (const float*)d_in[11];
  const float* attn_in  = (const float*)d_in[12];
  const float* attn_out = (const float*)d_in[13];
  const float* proj_w   = (const float*)d_in[14];
  const float* proj_b   = (const float*)d_in[15];
  float* out = (float*)d_out;

  char* ws = (char*)d_ws;
  size_t off = 0;
  auto alloc = [&](size_t bytes) -> void* {
    void* p = ws + off; off += (bytes + 255) & ~(size_t)255; return p;
  };
  unsigned short* W0cat = (unsigned short*)alloc((size_t)G4 * K0n * 2);
  unsigned short* W1cat = (unsigned short*)alloc((size_t)G4 * K1n * 2);
  unsigned short* Wq    = (unsigned short*)alloc((size_t)Hn * Hn * 2);
  unsigned short* Wo    = (unsigned short*)alloc((size_t)Hn * 2 * Hn * 2);
  unsigned short* Wp    = (unsigned short*)alloc((size_t)En * Hn * 2);
  unsigned short* Wemb  = (unsigned short*)alloc((size_t)Vn * En * 2);
  unsigned short* xbf   = (unsigned short*)alloc((size_t)Bn * Tn * En * 2);
  float* bias0 = (float*)alloc(G4 * 4);
  float* bias1 = (float*)alloc(G4 * 4);
  unsigned short* A0   = (unsigned short*)alloc((size_t)Bn * K0n * 2); // [x|feed|h0]
  unsigned short* A1   = (unsigned short*)alloc((size_t)Bn * K1n * 2); // [h0|h1]
  unsigned short* Aatt = (unsigned short*)alloc((size_t)Bn * K1n * 2); // [ctx|h1]
  float* c0    = (float*)alloc((size_t)Bn * Hn * 4);
  float* c1    = (float*)alloc((size_t)Bn * Hn * 4);
  float* gates = (float*)alloc((size_t)Bn * G4 * 4);
  float* qf    = (float*)alloc((size_t)Bn * Hn * 4);
  unsigned short* dec = (unsigned short*)alloc((size_t)Bn * Tn * Hn * 2);
  unsigned short* tmp = (unsigned short*)alloc((size_t)Bn * Tn * En * 2);

  // ---- weight conversion / fusion (runs every call; deterministic) ----
  k_build_w0<<<CDIV(G4 * K0n, 256), 256, 0, stream>>>(w_ih0, w_hh0, W0cat);
  k_build_w1<<<CDIV(G4 * K1n, 256), 256, 0, stream>>>(w_ih1, w_hh1, W1cat);
  k_cvt<<<CDIV(Hn * Hn, 256), 256, 0, stream>>>(attn_in, Wq, Hn * Hn);
  k_cvt<<<CDIV(Hn * 2 * Hn, 256), 256, 0, stream>>>(attn_out, Wo, Hn * 2 * Hn);
  k_cvt<<<CDIV(En * Hn, 256), 256, 0, stream>>>(proj_w, Wp, En * Hn);
  k_cvt<<<CDIV(Vn * En, 256), 256, 0, stream>>>(embed, Wemb, Vn * En);
  k_bias_sum<<<CDIV(G4, 256), 256, 0, stream>>>(b_ih0, b_hh0, bias0, G4);
  k_bias_sum<<<CDIV(G4, 256), 256, 0, stream>>>(b_ih1, b_hh1, bias1, G4);
  k_embed_gather<<<CDIV(Bn * Tn * En, 256), 256, 0, stream>>>(tokens, embed, xbf);
  // zero recurrent state: A0, A1, Aatt (bf16) + c0, c1 (f32)
  k_zero<<<CDIV(Bn * K0n / 2, 256), 256, 0, stream>>>((unsigned int*)A0, Bn * K0n / 2);
  k_zero<<<CDIV(Bn * K1n / 2, 256), 256, 0, stream>>>((unsigned int*)A1, Bn * K1n / 2);
  k_zero<<<CDIV(Bn * K1n / 2, 256), 256, 0, stream>>>((unsigned int*)Aatt, Bn * K1n / 2);
  k_zero<<<CDIV(Bn * Hn, 256), 256, 0, stream>>>((unsigned int*)c0, Bn * Hn);
  k_zero<<<CDIV(Bn * Hn, 256), 256, 0, stream>>>((unsigned int*)c1, Bn * Hn);

  // wave counts (exact multiples of 4 waves / 128-thread block)
  const int wav_g0  = (Bn / 16) * (G4 / 16);  // 2*128 = 256
  const int wav_g1  = wav_g0;
  const int wav_q   = (Bn / 16) * (Hn / 16);  // 2*32  = 64
  const int wav_att = wav_q;

  // ---- recurrent decode, T sequential steps ----
  for (int t = 0; t < Tn; ++t) {
    k_set_xt<<<CDIV(Bn * En, 256), 256, 0, stream>>>(xbf, A0, t);
    // layer 0 gates: [x|feed|h0] @ W0cat^T + bias0
    gemm_bf16_wmma<1, false><<<CDIV(wav_g0, 4), 128, 0, stream>>>(
        (const __bf16*)A0, K0n, (const __bf16*)W0cat, K0n, bias0,
        gates, G4, nullptr, 0, nullptr, 0, Bn, G4, K0n);
    k_lstm<<<CDIV(Bn * Hn, 256), 256, 0, stream>>>(
        gates, c0, A1, K1n, A0 + 1024, K0n);              // h0 -> A1[:,0:512], A0[:,1024:1536]
    // layer 1 gates: [h0|h1] @ W1cat^T + bias1
    gemm_bf16_wmma<1, false><<<CDIV(wav_g1, 4), 128, 0, stream>>>(
        (const __bf16*)A1, K1n, (const __bf16*)W1cat, K1n, bias1,
        gates, G4, nullptr, 0, nullptr, 0, Bn, G4, K1n);
    k_lstm<<<CDIV(Bn * Hn, 256), 256, 0, stream>>>(
        gates, c1, A1 + 512, K1n, Aatt + 512, K1n);       // h1 -> A1[:,512:], Aatt[:,512:]
    // q = h1 @ attn_in^T
    gemm_bf16_wmma<1, false><<<CDIV(wav_q, 4), 128, 0, stream>>>(
        (const __bf16*)(A1 + 512), K1n, (const __bf16*)Wq, Hn, nullptr,
        qf, Hn, nullptr, 0, nullptr, 0, Bn, Hn, Hn);
    // scores -> softmax -> ctx (ctx -> Aatt[:,0:512])
    k_attn<<<Bn, Sn, 0, stream>>>(enc, mask, qf, Aatt);
    // feed = tanh([ctx|h1] @ attn_out^T) -> A0[:,512:1024] and dec[:,t,:]
    gemm_bf16_wmma<1, true><<<CDIV(wav_att, 4), 128, 0, stream>>>(
        (const __bf16*)Aatt, K1n, (const __bf16*)Wo, K1n, nullptr,
        nullptr, 0, A0 + 512, K0n, dec + (size_t)t * Hn, Tn * Hn,
        Bn, Hn, K1n);
  }

  // ---- output projections ----
  // tmp = dec @ out_proj^T + b   (M=2048, N=512, K=512) -> bf16
  {
    const int waves = (Bn * Tn / 16) * (En / 16);          // 128*32 = 4096
    gemm_bf16_wmma<1, false><<<CDIV(waves, 4), 128, 0, stream>>>(
        (const __bf16*)dec, Hn, (const __bf16*)Wp, Hn, proj_b,
        nullptr, 0, tmp, En, nullptr, 0, Bn * Tn, En, Hn);
  }
  // feats = tmp @ embed^T  (M=2048, N=32000, K=512) -> f32 d_out; 16x64 per wave
  {
    const int waves = (Bn * Tn / 16) * (Vn / 64);          // 128*500 = 64000
    gemm_bf16_wmma<4, false><<<CDIV(waves, 4), 128, 0, stream>>>(
        (const __bf16*)tmp, En, (const __bf16*)Wemb, En, nullptr,
        out, Vn, nullptr, 0, nullptr, 0, Bn * Tn, Vn, En);
  }
}
